// DualGaussianNetwork_50946902065338
// MI455X (gfx1250) — matrix operand order repaired and checked by reading, not compiled, not run on hardware
//
#include <hip/hip_runtime.h>
#include <hip/hip_bf16.h>

// ---------------------------------------------------------------------------
// DualGaussianNetwork on gfx1250 (MI455X).
// All four GEMM stages run on v_wmma_f32_16x16x32_bf16 (f32 accum). Operands
// are pre-converted to bf16 (fused with the exact-f32 norm computation), so
// the GEMM inner loop is a pure byte mover: async global->LDS copies
// (ASYNCcnt) into double-buffered LDS, ds_load_b128 fragments, WMMA.
// Wave tile 64x64 (16 accumulators) -> 16 WMMA per 16 ds_load_b128.
// ---------------------------------------------------------------------------

typedef __attribute__((ext_vector_type(16))) __bf16 v16bf;
typedef __attribute__((ext_vector_type(8)))  __bf16 v8bf;
typedef __attribute__((ext_vector_type(8)))  float  v8f;
typedef __attribute__((ext_vector_type(4)))  int    v4i;

union FragAB { v16bf v; v8bf h[2]; };

#define BM 128
#define BN 256
#define BK 32
#define LDS_STRIDE 40   // 32 bf16 payload + 8 pad (80B rows): conflict-free b128 DS ops

enum { EPI_GAUSS = 0, EPI_RELU_BIAS = 1, EPI_BIAS = 2 };

#if defined(__has_builtin)
#  if __has_builtin(__builtin_amdgcn_global_load_async_to_lds_b128)
#    define ASYNC_COPY 1
#  endif
#  if __has_builtin(__builtin_amdgcn_s_wait_asynccnt)
#    define HAVE_WAIT_ASYNC 1
#  endif
#endif

__device__ __forceinline__ void copy16_g2l(const __bf16* __restrict__ g,
                                           __bf16* __restrict__ l)
{
#if ASYNC_COPY
    // Builtin signature (from compiler diagnostic): v4i addrspace(1)* src,
    // v4i addrspace(3)* lds, imm offset, imm cpol.
    __builtin_amdgcn_global_load_async_to_lds_b128(
        (__attribute__((address_space(1))) v4i*)(g),
        (__attribute__((address_space(3))) v4i*)(l),
        /*offset=*/0, /*cpol=*/0);
#else
    *reinterpret_cast<v8bf*>(l) = *reinterpret_cast<const v8bf*>(g);
#endif
}

__device__ __forceinline__ void wait_async_copies()
{
#if ASYNC_COPY
#  if HAVE_WAIT_ASYNC
    __builtin_amdgcn_s_wait_asynccnt(0);
#  else
    asm volatile("s_wait_asynccnt 0x0" ::: "memory");
#  endif
#endif
}

// Stage one BMxBK A-tile and BNxBK B-tile (bf16) into LDS buffers.
__device__ __forceinline__ void stage_tiles(const __bf16* __restrict__ A,
                                            const __bf16* __restrict__ W,
                                            __bf16 (*__restrict__ As)[LDS_STRIDE],
                                            __bf16 (*__restrict__ Bs)[LDS_STRIDE],
                                            int rowBase, int colBase,
                                            int K, int k0, int tid)
{
    // A: 128x32 bf16 = 512 x 16B chunks; 2 per thread.
    #pragma unroll
    for (int i = 0; i < 2; ++i) {
        const int c  = tid + i * 256;
        const int r  = c >> 2;
        const int c4 = c & 3;
        copy16_g2l(A + (size_t)(rowBase + r) * K + k0 + c4 * 8, &As[r][c4 * 8]);
    }
    // B: 256x32 bf16 = 1024 x 16B chunks; 4 per thread.
    #pragma unroll
    for (int i = 0; i < 4; ++i) {
        const int c  = tid + i * 256;
        const int r  = c >> 2;
        const int c4 = c & 3;
        copy16_g2l(W + (size_t)(colBase + r) * K + k0 + c4 * 8, &Bs[r][c4 * 8]);
    }
}

// C = A(MxK bf16) * W(NxK bf16)^T with fused epilogue.
//   EPI_GAUSS:     r = exp(-max(rowAux[i] - 2*acc + colAux[j], 0) / sig[j]^2)
//   EPI_RELU_BIAS: r = max(acc + colAux[j], 0)
//   EPI_BIAS:      r = acc + colAux[j]
// Writes r to outF (f32) and/or outB (bf16) when non-null.
template <int EPI>
__global__ __launch_bounds__(256)
void gemm_bf16_wmma(const __bf16* __restrict__ A,
                    const __bf16* __restrict__ W,
                    const float* __restrict__ rowAux,
                    const float* __restrict__ colAux,
                    const float* __restrict__ sig,
                    float* __restrict__ outF,
                    __bf16* __restrict__ outB,
                    int M, int N, int K)
{
    __shared__ __align__(16) __bf16 As[2][BM][LDS_STRIDE];  // 20 KB
    __shared__ __align__(16) __bf16 Bs[2][BN][LDS_STRIDE];  // 40 KB

    const int tid  = threadIdx.x;
    const int lane = tid & 31;
    const int wave = tid >> 5;
    const int wm   = wave >> 2;     // 0..1 : 64-row strip
    const int wn   = wave & 3;      // 0..3 : 64-col strip
    const int lo   = lane & 15;
    const int hi   = lane >> 4;

    const int rowBase = blockIdx.y * BM;
    const int colBase = blockIdx.x * BN;

    v8f acc[4][4];
    const v8f vzero = {0.f, 0.f, 0.f, 0.f, 0.f, 0.f, 0.f, 0.f};
    #pragma unroll
    for (int tm = 0; tm < 4; ++tm)
        #pragma unroll
        for (int tn = 0; tn < 4; ++tn)
            acc[tm][tn] = vzero;

    const int nk = K / BK;
    // Prologue: fill buffer 0.
    stage_tiles(A, W, As[0], Bs[0], rowBase, colBase, K, 0, tid);

    for (int ks = 0; ks < nk; ++ks) {
        const int buf = ks & 1;
        wait_async_copies();
        __syncthreads();            // buf ready; previous buffer's readers done

        if (ks + 1 < nk)            // prefetch next tile into the other buffer
            stage_tiles(A, W, As[buf ^ 1], Bs[buf ^ 1],
                        rowBase, colBase, K, (ks + 1) * BK, tid);

        // A frag (16x32): lane=row lo; elems 0..7 = K[hi*8..], 8..15 = K[16+hi*8..]
        FragAB af[4];
        #pragma unroll
        for (int tm = 0; tm < 4; ++tm) {
            const int mrow = wm * 64 + tm * 16 + lo;
            af[tm].h[0] = *reinterpret_cast<const v8bf*>(&As[buf][mrow][hi * 8]);
            af[tm].h[1] = *reinterpret_cast<const v8bf*>(&As[buf][mrow][16 + hi * 8]);
        }
        // B frag (32x16 = W^T tile): lane=col lo; elems 0..7 = K[hi*8..], 8..15 = K[16+hi*8..]
        FragAB bfr[4];
        #pragma unroll
        for (int tn = 0; tn < 4; ++tn) {
            const int ncol = wn * 64 + tn * 16 + lo;
            bfr[tn].h[0] = *reinterpret_cast<const v8bf*>(&Bs[buf][ncol][hi * 8]);
            bfr[tn].h[1] = *reinterpret_cast<const v8bf*>(&Bs[buf][ncol][16 + hi * 8]);
        }

        #pragma unroll
        for (int tm = 0; tm < 4; ++tm)
            #pragma unroll
            for (int tn = 0; tn < 4; ++tn)
                acc[tm][tn] = __builtin_amdgcn_wmma_f32_16x16x32_bf16(
                    false, af[tm].v, false, bfr[tn].v,
                    (short)0, acc[tm][tn], false, false);
    }

    // Epilogue: C/D layout -> VGPR e, lane l = (M = e + 8*hi, N = lo).
    #pragma unroll
    for (int tm = 0; tm < 4; ++tm) {
        #pragma unroll
        for (int tn = 0; tn < 4; ++tn) {
            const int gcol = colBase + wn * 64 + tn * 16 + lo;
            const float caux = colAux[gcol];
            float inv_s2 = 0.f;
            if (EPI == EPI_GAUSS) {
                const float s = sig[gcol];
                inv_s2 = 1.0f / (s * s);
            }
            #pragma unroll
            for (int e = 0; e < 8; ++e) {
                const int grow = rowBase + wm * 64 + tm * 16 + hi * 8 + e;
                const float v = acc[tm][tn][e];
                float r;
                if (EPI == EPI_GAUSS) {
                    float d2 = rowAux[grow] - 2.0f * v + caux;
                    d2 = fmaxf(d2, 0.0f);
                    r = __expf(-d2 * inv_s2);
                } else if (EPI == EPI_RELU_BIAS) {
                    r = fmaxf(v + caux, 0.0f);
                } else {
                    r = v + caux;
                }
                const size_t idx = (size_t)grow * N + gcol;
                if (outF) outF[idx] = r;
                if (outB) outB[idx] = (__bf16)r;
            }
        }
    }
}

// f32 -> bf16 row conversion fused with exact-f32 squared row norm.
__global__ __launch_bounds__(256)
void convert_norm_kernel(const float* __restrict__ A, __bf16* __restrict__ outB,
                         float* __restrict__ norm, int K)
{
    const int row = blockIdx.x;
    const float* src = A + (size_t)row * K;
    __bf16* dst = outB + (size_t)row * K;
    float s = 0.f;
    for (int i = threadIdx.x; i < K; i += 256) {
        const float v = src[i];
        s += v * v;
        dst[i] = (__bf16)v;
    }
    if (norm != nullptr) {
        #pragma unroll
        for (int off = 16; off > 0; off >>= 1)
            s += __shfl_down(s, off, 32);
        __shared__ float partial[8];
        if ((threadIdx.x & 31) == 0) partial[threadIdx.x >> 5] = s;
        __syncthreads();
        if (threadIdx.x == 0) {
            float t = 0.f;
            #pragma unroll
            for (int w = 0; w < 8; ++w) t += partial[w];
            norm[row] = t;
        }
    }
}

// Squared row norm of a bf16 matrix (f32 accumulation).
__global__ __launch_bounds__(256)
void rownorm_bf16_kernel(const __bf16* __restrict__ A, float* __restrict__ out, int K)
{
    const int row = blockIdx.x;
    const __bf16* r = A + (size_t)row * K;
    float s = 0.f;
    for (int i = threadIdx.x; i < K; i += 256) {
        const float v = (float)r[i];
        s += v * v;
    }
    #pragma unroll
    for (int off = 16; off > 0; off >>= 1)
        s += __shfl_down(s, off, 32);
    __shared__ float partial[8];
    if ((threadIdx.x & 31) == 0) partial[threadIdx.x >> 5] = s;
    __syncthreads();
    if (threadIdx.x == 0) {
        float t = 0.f;
        #pragma unroll
        for (int w = 0; w < 8; ++w) t += partial[w];
        out[row] = t;
    }
}

extern "C" void kernel_launch(void* const* d_in, const int* in_sizes, int n_in,
                              void* d_out, int out_size, void* d_ws, size_t ws_size,
                              hipStream_t stream)
{
    (void)in_sizes; (void)n_in; (void)out_size; (void)ws_size;
    const int B = 8192, IN = 512, HID = 1024, OUT = 512, RH = 2048;

    const float* x  = (const float*)d_in[0];
    const float* c1 = (const float*)d_in[1];
    const float* s1 = (const float*)d_in[2];
    const float* c2 = (const float*)d_in[3];
    const float* s2 = (const float*)d_in[4];
    const float* w1 = (const float*)d_in[5];
    const float* b1 = (const float*)d_in[6];
    const float* w2 = (const float*)d_in[7];
    const float* b2 = (const float*)d_in[8];

    float* features = (float*)d_out;                  // (B, OUT) f32
    float* enhanced = features + (size_t)B * OUT;     // (B, OUT) f32

    // Workspace layout (~70MB; intermediates bf16, all L2-resident).
    char* w = (char*)d_ws;
    __bf16* xb   = (__bf16*)w;  w += (size_t)B   * IN  * 2;
    __bf16* c1b  = (__bf16*)w;  w += (size_t)HID * IN  * 2;
    __bf16* c2b  = (__bf16*)w;  w += (size_t)OUT * HID * 2;
    __bf16* w1b  = (__bf16*)w;  w += (size_t)RH  * OUT * 2;
    __bf16* w2b  = (__bf16*)w;  w += (size_t)OUT * RH  * 2;
    __bf16* hb   = (__bf16*)w;  w += (size_t)B   * HID * 2;
    __bf16* ftb  = (__bf16*)w;  w += (size_t)B   * OUT * 2;
    __bf16* hdb  = (__bf16*)w;  w += (size_t)B   * RH  * 2;
    float* xnorm  = (float*)w;  w += (size_t)B   * 4;
    float* c1norm = (float*)w;  w += (size_t)HID * 4;
    float* c2norm = (float*)w;  w += (size_t)OUT * 4;
    float* hnorm  = (float*)w;

    // bf16 conversion fused with exact f32 squared norms.
    convert_norm_kernel<<<B,   256, 0, stream>>>(x,  xb,  xnorm,  IN);
    convert_norm_kernel<<<HID, 256, 0, stream>>>(c1, c1b, c1norm, IN);
    convert_norm_kernel<<<OUT, 256, 0, stream>>>(c2, c2b, c2norm, HID);
    convert_norm_kernel<<<RH,  256, 0, stream>>>(w1, w1b, nullptr, OUT);
    convert_norm_kernel<<<OUT, 256, 0, stream>>>(w2, w2b, nullptr, RH);

    // Stage 1: h = exp(-max(||x||^2 - 2 x.c1 + ||c1||^2, 0)/s1^2)   (bf16 out)
    gemm_bf16_wmma<EPI_GAUSS><<<dim3(HID / BN, B / BM), 256, 0, stream>>>(
        xb, c1b, xnorm, c1norm, s1, nullptr, hb, B, HID, IN);

    rownorm_bf16_kernel<<<B, 256, 0, stream>>>(hb, hnorm, HID);

    // Stage 2: features = exp(-d2(h, c2)/s2^2)   (f32 to d_out + bf16 to ws)
    gemm_bf16_wmma<EPI_GAUSS><<<dim3(OUT / BN, B / BM), 256, 0, stream>>>(
        hb, c2b, hnorm, c2norm, s2, features, ftb, B, OUT, HID);

    // Stage 3: hidden = relu(features . w1^T + b1)   (bf16 out)
    gemm_bf16_wmma<EPI_RELU_BIAS><<<dim3(RH / BN, B / BM), 256, 0, stream>>>(
        ftb, w1b, nullptr, b1, nullptr, nullptr, hdb, B, RH, OUT);

    // Stage 4: enhanced = hidden . w2^T + b2   (f32 to d_out)
    gemm_bf16_wmma<EPI_BIAS><<<dim3(OUT / BN, B / BM), 256, 0, stream>>>(
        hdb, w2b, nullptr, b2, nullptr, enhanced, nullptr, B, OUT, RH);
}
